// MatchingModel_GATv2SinkhornTopK_47141561041003
// MI455X (gfx1250) — matrix operand order repaired and checked by reading, not compiled, not run on hardware
//
#include <hip/hip_runtime.h>
#include <hip/hip_bf16.h>
#include <math.h>

// ---------------- problem constants ----------------
#define BB   64
#define NN   256
#define FIN  7
#define HID  64
#define OUTD 32
#define DEG  32
#define NT   (BB * NN)            // 16384 nodes per side
#define EPG  (NN * DEG)           // 8192 edges per graph
#define ETOT (EPG + NN)           // + self loops
#define ESIDE ((size_t)BB * EPG)  // 524288 edges per side
#define SIMN (NN * NN)            // 65536 per batch
#define EPSV 1e-5f

typedef _Float16 v16h __attribute__((ext_vector_type(16)));
typedef _Float16 v8h  __attribute__((ext_vector_type(8)));
typedef float    v8f  __attribute__((ext_vector_type(8)));

// float atomic-max on LDS via signed/unsigned bit trick (init to -inf)
__device__ inline void atomicMaxFloatLds(float* addr, float val) {
  if (val >= 0.0f) atomicMax((int*)addr, __float_as_int(val));
  else             atomicMin((unsigned int*)addr, __float_as_uint(val));
}

__device__ inline float softplusf(float x) {
  if (x > 20.0f)  return x;
  if (x < -20.0f) return __expf(x);
  return log1pf(__expf(x));
}

// ---------------- K1: layer-1 node transforms (IN=7 -> HID=64) ----------------
__global__ void node_l1_kernel(const float* __restrict__ x1, const float* __restrict__ x2,
                               const float* __restrict__ Wl, const float* __restrict__ Wr,
                               const float* __restrict__ bl, const float* __restrict__ br,
                               float* __restrict__ xl, float* __restrict__ xr) {
  __shared__ float sWl[FIN * HID], sWr[FIN * HID], sbl[HID], sbr[HID];
  int tid = threadIdx.x;
  for (int i = tid; i < FIN * HID; i += 256) { sWl[i] = Wl[i]; sWr[i] = Wr[i]; }
  if (tid < HID) { sbl[tid] = bl[tid]; sbr[tid] = br[tid]; }
  __syncthreads();

  int node = blockIdx.x * 256 + tid;                 // 0 .. 2*NT-1
  const float* x = (node < NT) ? (x1 + (size_t)node * FIN)
                               : (x2 + (size_t)(node - NT) * FIN);
  float xv[FIN];
#pragma unroll
  for (int k = 0; k < FIN; ++k) xv[k] = x[k];

  float* ol = xl + (size_t)node * HID;
  float* orr = xr + (size_t)node * HID;
#pragma unroll 4
  for (int d = 0; d < HID; ++d) {
    float al = sbl[d], ar = sbr[d];
#pragma unroll
    for (int k = 0; k < FIN; ++k) {
      al = fmaf(xv[k], sWl[k * HID + d], al);
      ar = fmaf(xv[k], sWr[k * HID + d], ar);
    }
    ol[d] = al; orr[d] = ar;
  }
}

// ---------------- K3: layer-2 node transforms (HID=64 -> OUT=32) ----------------
__global__ void node_l2_kernel(const float* __restrict__ h,
                               const float* __restrict__ Wl, const float* __restrict__ Wr,
                               const float* __restrict__ bl, const float* __restrict__ br,
                               float* __restrict__ xl, float* __restrict__ xr) {
  __shared__ float sWl[HID * OUTD], sWr[HID * OUTD], sbl[OUTD], sbr[OUTD];
  int tid = threadIdx.x;
  for (int i = tid; i < HID * OUTD; i += 256) { sWl[i] = Wl[i]; sWr[i] = Wr[i]; }
  if (tid < OUTD) { sbl[tid] = bl[tid]; sbr[tid] = br[tid]; }
  __syncthreads();

  int node = blockIdx.x * 256 + tid;
  const float4* hp = (const float4*)(h + (size_t)node * HID);
  float hv[HID];
#pragma unroll
  for (int q = 0; q < HID / 4; ++q) {
    float4 t = hp[q];
    hv[4 * q + 0] = t.x; hv[4 * q + 1] = t.y; hv[4 * q + 2] = t.z; hv[4 * q + 3] = t.w;
  }
  float* ol = xl + (size_t)node * OUTD;
  float* orr = xr + (size_t)node * OUTD;
  for (int d = 0; d < OUTD; ++d) {
    float al = sbl[d], ar = sbr[d];
#pragma unroll
    for (int k = 0; k < HID; ++k) {
      al = fmaf(hv[k], sWl[k * OUTD + d], al);
      ar = fmaf(hv[k], sWr[k * OUTD + d], ar);
    }
    ol[d] = al; orr[d] = ar;
  }
}

// ---------------- K2/K4: GATv2 edge softmax + aggregation, one graph per WGP ----
// L1=true : D=64, output = relu(agg + bias) as f32
// L1=false: D=32, output = (agg + bias) as f16 (feeds WMMA sim kernel)
template <int D, int PAD, bool L1>
__global__ void edge_agg_kernel(const float* __restrict__ xl_g, const float* __restrict__ xr_g,
                                const int* __restrict__ edge1, const int* __restrict__ edge2,
                                const float* __restrict__ att, const float* __restrict__ bias,
                                float* __restrict__ out_f32, _Float16* __restrict__ out_f16) {
  __shared__ float s_xl[NN * PAD];
  __shared__ float s_xr[NN * PAD];
  __shared__ float s_acc[NN * PAD];
  __shared__ float s_esc[ETOT];
  __shared__ float s_smax[NN];
  __shared__ float s_den[NN];
  __shared__ float s_att[D];

  const int tid  = threadIdx.x;
  const int gb   = blockIdx.x & 63;          // graph within side
  const int side = blockIdx.x >> 6;          // 0: graph set 1, 1: graph set 2
  const int* eb  = side ? edge2 : edge1;
  const int* src = eb + (size_t)gb * EPG;
  const int* dst = eb + ESIDE + (size_t)gb * EPG;
  const int node_base = side * NT + gb * NN; // row into xl_g/xr_g/out
  const int goff = gb * NN;                  // global->local node id offset

  // load this graph's transformed features into LDS (padded rows: no bank conflicts)
  {
    const float* pl = xl_g + (size_t)(node_base + tid) * D;
    const float* pr = xr_g + (size_t)(node_base + tid) * D;
#pragma unroll 4
    for (int k = 0; k < D; ++k) {
      s_xl[tid * PAD + k]  = pl[k];
      s_xr[tid * PAD + k]  = pr[k];
      s_acc[tid * PAD + k] = 0.0f;
    }
    s_smax[tid] = -INFINITY;
    s_den[tid]  = 0.0f;
    if (tid < D) s_att[tid] = att[tid];
  }
  __syncthreads();

  // pass A: scores + segment max
  for (int e = tid; e < ETOT; e += 256) {
    int sL, tL;
    if (e < EPG) { sL = src[e] - goff; tL = dst[e] - goff; }
    else         { sL = tL = e - EPG; }                       // self loop
    float sc = 0.0f;
#pragma unroll 4
    for (int k = 0; k < D; ++k) {
      float v = s_xl[sL * PAD + k] + s_xr[tL * PAD + k];
      v = (v > 0.0f) ? v : 0.2f * v;                          // leaky_relu 0.2
      sc = fmaf(v, s_att[k], sc);
    }
    s_esc[e] = sc;
    atomicMaxFloatLds(&s_smax[tL], sc);
  }
  __syncthreads();

  // pass B: exp + segment sum
  for (int e = tid; e < ETOT; e += 256) {
    int tL = (e < EPG) ? (dst[e] - goff) : (e - EPG);
    float ex = __expf(s_esc[e] - s_smax[tL]);
    s_esc[e] = ex;
    atomicAdd(&s_den[tL], ex);
  }
  __syncthreads();

  // pass C: weighted scatter of xl[src]
  for (int e = tid; e < ETOT; e += 256) {
    int sL, tL;
    if (e < EPG) { sL = src[e] - goff; tL = dst[e] - goff; }
    else         { sL = tL = e - EPG; }
    float alpha = s_esc[e] / s_den[tL];
#pragma unroll 4
    for (int k = 0; k < D; ++k)
      atomicAdd(&s_acc[tL * PAD + k], alpha * s_xl[sL * PAD + k]);
  }
  __syncthreads();

  // write out
  if (L1) {
    float* o = out_f32 + (size_t)(node_base + tid) * D;
#pragma unroll 4
    for (int k = 0; k < D; ++k) {
      float v = s_acc[tid * PAD + k] + bias[k];
      o[k] = fmaxf(v, 0.0f);                                  // relu between layers
    }
  } else {
    _Float16* o = out_f16 + (size_t)(node_base + tid) * D;
#pragma unroll 4
    for (int k = 0; k < D; ++k)
      o[k] = (_Float16)(s_acc[tid * PAD + k] + bias[k]);
  }
}

// ---------------- K5: sim = h1 @ h2^T via v_wmma_f32_16x16x32_f16 --------------
// one wave32 per 16x16 output tile; K=32 == exactly one WMMA
__global__ void sim_wmma_kernel(const _Float16* __restrict__ h16, float* __restrict__ sim) {
  const int lane   = threadIdx.x & 31;
  const int wave   = threadIdx.x >> 5;
  const int tileId = blockIdx.x * 8 + wave;      // 64*256 tiles total
  const int b  = tileId >> 8;
  const int tt = tileId & 255;
  const int tm = tt >> 4, tn = tt & 15;

  // A: 16x32 f16 — lane<16 holds M=lane, K 0..15; lane>=16 same M, K 16..31
  const _Float16* pa = h16 + ((size_t)(b * NN + tm * 16 + (lane & 15))) * OUTD;
  const int k0 = (lane >> 4) * 16;
  v8h a0 = *(const v8h*)(pa + k0);
  v8h a1 = *(const v8h*)(pa + k0 + 8);
  v16h A;
#pragma unroll
  for (int i = 0; i < 8; ++i) { A[i] = a0[i]; A[i + 8] = a1[i]; }

  // B: 32x16 f16 — lane<16: col N=lane, K 0..15; lane>=16: col N=lane-16, K 16..31
  const _Float16* pb = h16 + (size_t)NT * OUTD
                     + ((size_t)(b * NN + tn * 16 + (lane & 15))) * OUTD
                     + (lane >> 4) * 16;
  v16h Bv = *(const v16h*)pb;

  v8f c = {};
  v8f d = __builtin_amdgcn_wmma_f32_16x16x32_f16(false, A, false, Bv,
                                                 (short)0, c, false, false);

  float* so = sim + (size_t)b * SIMN;
  const int n = tn * 16 + (lane & 15);
  const int mbase = tm * 16 + ((lane < 16) ? 0 : 8);
#pragma unroll
  for (int r = 0; r < 8; ++r)
    so[(size_t)(mbase + r) * NN + n] = d[r];
}

// ---------------- K6: InstanceNorm + collapsed Sinkhorn top-k ------------------
// d0_i = 2*s_i - mn - mx ; row-norm is a pure fn of d; col-norm shifts d by a
// per-batch scalar. 10 iterations -> 4 scalar updates + 1 final pass.
__global__ void sinkhorn_kernel(const float* __restrict__ sim,
                                const float* __restrict__ gamma,
                                const float* __restrict__ beta,
                                float* __restrict__ out) {
  __shared__ float rA[256], rB[256], rC[256], rD[256];
  __shared__ float bc[6];   // P, Q, R, D, LSEb
  const int b = blockIdx.x, tid = threadIdx.x;
  const float* base = sim + (size_t)b * SIMN;

  // pass 0: mean / var / min / max
  float sum = 0.f, ss = 0.f, mn = INFINITY, mx = -INFINITY;
  for (int i = tid; i < SIMN; i += 256) {
    float v = base[i];
    sum += v; ss = fmaf(v, v, ss);
    mn = fminf(mn, v); mx = fmaxf(mx, v);
  }
  rA[tid] = sum; rB[tid] = ss; rC[tid] = mn; rD[tid] = mx;
  __syncthreads();
  for (int s = 128; s > 0; s >>= 1) {
    if (tid < s) {
      rA[tid] += rA[tid + s]; rB[tid] += rB[tid + s];
      rC[tid] = fminf(rC[tid], rC[tid + s]);
      rD[tid] = fmaxf(rD[tid], rD[tid + s]);
    }
    __syncthreads();
  }
  if (tid == 0) {
    const float inv_n = 1.0f / (float)SIMN;
    float mu  = rA[0] * inv_n;
    float var = fmaxf(rB[0] * inv_n - mu * mu, 0.0f);
    float rs  = rsqrtf(var + EPSV);
    float g = gamma[0], bt = beta[0];
    float a = g * rs, off = bt - a * mu;
    float smn = a * rC[0] + off, smx = a * rD[0] + off;
    if (a < 0.0f) { float t = smn; smn = smx; smx = t; }
    bc[0] = 2.0f * a;                       // P : d = P*sim + Q + D
    bc[1] = 2.0f * off - smn - smx;         // Q
    bc[2] = smx - smn;                      // R : d0 range is [-R, R]
    bc[3] = 0.0f;                           // D : accumulated column shift
  }
  __syncthreads();

  const float P = bc[0], Q = bc[1], R = bc[2];
  const float logk_nk = logf((float)NN / (float)(SIMN - NN));  // log(k/(n-k))

  // 4 (row,col) Sinkhorn pairs -> scalar shift updates
  for (int t = 0; t < 4; ++t) {
    const float Dv = bc[3];
    const float Ma = -softplusf(Dv - R);       // max of -softplus(d)
    const float Mb = -softplusf(-(Dv + R));    // max of d - softplus(d)
    float Sa = 0.f, Sb = 0.f;
    for (int i = tid; i < SIMN; i += 256) {
      float d  = fmaf(P, base[i], Q) + Dv;
      float sp = softplusf(d);
      Sa += __expf(-sp - Ma);
      Sb += __expf(d - sp - Mb);
    }
    rA[tid] = Sa; rB[tid] = Sb;
    __syncthreads();
    for (int s = 128; s > 0; s >>= 1) {
      if (tid < s) { rA[tid] += rA[tid + s]; rB[tid] += rB[tid + s]; }
      __syncthreads();
    }
    if (tid == 0) {
      float c = (Ma + logf(rA[0])) - (Mb + logf(rB[0])) + logk_nk;
      bc[3] = bc[3] + c;
    }
    __syncthreads();
  }

  // final column step + clip(exp(b)*n)
  {
    const float Dv = bc[3];
    const float Mb = -softplusf(-(Dv + R));
    float Sb = 0.f;
    for (int i = tid; i < SIMN; i += 256) {
      float d = fmaf(P, base[i], Q) + Dv;
      Sb += __expf(d - softplusf(d) - Mb);
    }
    rB[tid] = Sb;
    __syncthreads();
    for (int s = 128; s > 0; s >>= 1) {
      if (tid < s) rB[tid] += rB[tid + s];
      __syncthreads();
    }
    if (tid == 0) bc[4] = Mb + logf(rB[0]);    // LSE of (d - softplus(d))
    __syncthreads();
    const float LSEb = bc[4];
    float* o = out + (size_t)b * SIMN;
    for (int i = tid; i < SIMN; i += 256) {
      float d = fmaf(P, base[i], Q) + Dv;
      float v = __expf(d - softplusf(d) - LSEb) * (float)NN;  // exp(b)*n
      o[i] = fminf(fmaxf(v, 0.0f), 1.0f);
    }
  }
}

// ---------------- host launch --------------------------------------------------
extern "C" void kernel_launch(void* const* d_in, const int* in_sizes, int n_in,
                              void* d_out, int out_size, void* d_ws, size_t ws_size,
                              hipStream_t stream) {
  const float* x1    = (const float*)d_in[0];
  const float* x2    = (const float*)d_in[1];
  const int*   edge1 = (const int*)d_in[2];
  const int*   edge2 = (const int*)d_in[3];
  const float* Wl1   = (const float*)d_in[4];
  const float* Wr1   = (const float*)d_in[5];
  const float* bl1   = (const float*)d_in[6];
  const float* br1   = (const float*)d_in[7];
  const float* att1  = (const float*)d_in[8];
  const float* bias1 = (const float*)d_in[9];
  const float* Wl2   = (const float*)d_in[10];
  const float* Wr2   = (const float*)d_in[11];
  const float* bl2   = (const float*)d_in[12];
  const float* br2   = (const float*)d_in[13];
  const float* att2  = (const float*)d_in[14];
  const float* bias2 = (const float*)d_in[15];
  const float* gamma = (const float*)d_in[16];
  const float* beta  = (const float*)d_in[17];
  float* out = (float*)d_out;

  // workspace layout (regions reused once dead); peak need = 32 MB
  char* ws = (char*)d_ws;
  const size_t MB = 1024 * 1024;
  float*     xl1  = (float*)(ws + 0 * MB);       // 8 MB  (2*NT x 64)
  float*     xr1  = (float*)(ws + 8 * MB);       // 8 MB
  float*     h1   = (float*)(ws + 16 * MB);      // 8 MB
  float*     xl2  = (float*)(ws + 0 * MB);       // 4 MB  (reuses xl1)
  float*     xr2  = (float*)(ws + 4 * MB);       // 4 MB  (reuses xl1 tail)
  _Float16*  h16  = (_Float16*)(ws + 8 * MB);    // 2 MB  (reuses xr1)
  float*     sim  = (float*)(ws + 16 * MB);      // 16 MB (reuses h1)

  node_l1_kernel<<<2 * NT / 256, 256, 0, stream>>>(x1, x2, Wl1, Wr1, bl1, br1, xl1, xr1);
  edge_agg_kernel<HID, HID + 1, true><<<2 * BB, 256, 0, stream>>>(
      xl1, xr1, edge1, edge2, att1, bias1, h1, nullptr);
  node_l2_kernel<<<2 * NT / 256, 256, 0, stream>>>(h1, Wl2, Wr2, bl2, br2, xl2, xr2);
  edge_agg_kernel<OUTD, OUTD + 1, false><<<2 * BB, 256, 0, stream>>>(
      xl2, xr2, edge1, edge2, att2, bias2, nullptr, h16);
  sim_wmma_kernel<<<(BB * 256) / 8, 256, 0, stream>>>(h16, sim);
  sinkhorn_kernel<<<BB, 256, 0, stream>>>(sim, gamma, beta, out);
}
